// BayesianNN_27298812133619
// MI455X (gfx1250) — matrix operand order repaired and checked by reading, not compiled, not run on hardware
//
#include <hip/hip_runtime.h>
#include <stdint.h>

#define MM 256
#define DD 7686
#define HH 256
#define OO 4
#define NTOT (DD + HH + OO)   // 7946
#define HO (HH + OO)          // 260

typedef __bf16 bf16_t;
typedef __bf16    v16bf __attribute__((ext_vector_type(16)));
typedef float     v8f   __attribute__((ext_vector_type(8)));
typedef float     v4f_u __attribute__((ext_vector_type(4), aligned(4)));
typedef unsigned  v4u   __attribute__((ext_vector_type(4)));
typedef int       v8i   __attribute__((ext_vector_type(8)));

// ---- unguarded: 8 contiguous fp32 -> half of a v16bf (b128 loads, 4B align ok)
__device__ __forceinline__ void cvt8f(const float* __restrict__ p, v16bf& v, int base) {
  v4f_u a = *(const v4f_u*)(p);
  v4f_u b = *(const v4f_u*)(p + 4);
  v[base + 0] = (bf16_t)a.x; v[base + 1] = (bf16_t)a.y;
  v[base + 2] = (bf16_t)a.z; v[base + 3] = (bf16_t)a.w;
  v[base + 4] = (bf16_t)b.x; v[base + 5] = (bf16_t)b.y;
  v[base + 6] = (bf16_t)b.z; v[base + 7] = (bf16_t)b.w;
}
// ---- guarded (K tail only)
__device__ __forceinline__ void cvt8g(const float* __restrict__ p, int rem,
                                      v16bf& v, int base) {
#pragma unroll
  for (int i = 0; i < 8; ++i) v[base + i] = (bf16_t)((i < rem) ? p[i] : 0.0f);
}

// A fragment (16-bit A 16x32, ISA 7.12.2): lane={half,l16}; row M=l16;
//   elems 0..7 = K(k0+half*8..), elems 8..15 = K(k0+16+half*8..)
__device__ __forceinline__ void frag_a(const float* pa, int k0, int half, v16bf& a) {
  cvt8f(pa + k0 + half * 8, a, 0);
  cvt8f(pa + k0 + 16 + half * 8, a, 8);
}
// B fragment: col N=l16; elems 0..7 = K(k0+half*8..), elems 8..15 = K(k0+16+half*8..)
__device__ __forceinline__ void frag_b(const float* pb, int k0, int half, v16bf& b) {
  cvt8f(pb + k0 + half * 8, b, 0);
  cvt8f(pb + k0 + 16 + half * 8, b, 8);
}

struct Acc2 { v8f a0, a1; };

// per-wave 16x32 tile of D[m,n] = sum_k A[m,k]*B[n,k] (both row-major along K).
// OOB columns: clamp B row pointer (results discarded at store), no loop branch.
__device__ __forceinline__ Acc2 wmma_rowrow2(const float* __restrict__ A, int lda,
                                             const float* __restrict__ B, int ldb,
                                             int Klen, int mtile, int ntile, int nmax) {
  const int lane = threadIdx.x & 31;
  const int l16  = lane & 15;
  const int half = lane >> 4;
  const float* pa = A + (size_t)(mtile + l16) * lda;
  int nr0 = ntile + l16;      if (nr0 >= nmax) nr0 = nmax - 1;
  int nr1 = ntile + 16 + l16; if (nr1 >= nmax) nr1 = nmax - 1;
  const float* pb0 = B + (size_t)nr0 * ldb;
  const float* pb1 = B + (size_t)nr1 * ldb;
  v8f acc0 = {}, acc1 = {};
  const int KTF = Klen / 32;                 // full 32-K chunks (even: unroll x2)
  for (int kt = 0; kt < KTF; kt += 2) {
    const int k0 = kt * 32;
    v16bf a0, a1, b00, b10, b01, b11;
    frag_a(pa, k0, half, a0);
    frag_b(pb0, k0, half, b00);
    frag_b(pb1, k0, half, b10);
    frag_a(pa, k0 + 32, half, a1);
    frag_b(pb0, k0 + 32, half, b01);
    frag_b(pb1, k0 + 32, half, b11);
    acc0 = __builtin_amdgcn_wmma_f32_16x16x32_bf16(false, a0, false, b00, (short)0, acc0, false, false);
    acc1 = __builtin_amdgcn_wmma_f32_16x16x32_bf16(false, a0, false, b10, (short)0, acc1, false, false);
    acc0 = __builtin_amdgcn_wmma_f32_16x16x32_bf16(false, a1, false, b01, (short)0, acc0, false, false);
    acc1 = __builtin_amdgcn_wmma_f32_16x16x32_bf16(false, a1, false, b11, (short)0, acc1, false, false);
  }
  if (KTF * 32 < Klen) {                     // guarded K tail (rem = Klen % 32)
    const int k0 = KTF * 32;
    v16bf a, b0, b1;
    cvt8g(pa + k0 + half * 8,      Klen - (k0 + half * 8),      a, 0);
    cvt8g(pa + k0 + 16 + half * 8, Klen - (k0 + 16 + half * 8), a, 8);
    const int kb = k0 + half * 8;
    cvt8g(pb0 + kb,      Klen - kb,      b0, 0);
    cvt8g(pb0 + kb + 16, Klen - kb - 16, b0, 8);
    cvt8g(pb1 + kb,      Klen - kb,      b1, 0);
    cvt8g(pb1 + kb + 16, Klen - kb - 16, b1, 8);
    acc0 = __builtin_amdgcn_wmma_f32_16x16x32_bf16(false, a, false, b0, (short)0, acc0, false, false);
    acc1 = __builtin_amdgcn_wmma_f32_16x16x32_bf16(false, a, false, b1, (short)0, acc1, false, false);
  }
  return Acc2{acc0, acc1};
}

// ================= kernel 1: Q/K/V = X @ W.T + b  (bf16 WMMA) =================
__global__ __launch_bounds__(256) void qkv_gemm(
    const float* __restrict__ X,
    const float* __restrict__ Wq, const float* __restrict__ bq,
    const float* __restrict__ Wk, const float* __restrict__ bk,
    const float* __restrict__ Wv, const float* __restrict__ bv,
    float* __restrict__ Qo, float* __restrict__ Ko, float* __restrict__ Vo) {
  const float* Wm; const float* bb; float* outp;
  if (blockIdx.z == 0)      { Wm = Wq; bb = bq; outp = Qo; }
  else if (blockIdx.z == 1) { Wm = Wk; bb = bk; outp = Ko; }
  else                      { Wm = Wv; bb = bv; outp = Vo; }
  const int wv    = threadIdx.x >> 5;
  const int ntile = blockIdx.x * 256 + wv * 32;   // 16x32 output per wave
  const int mtile = blockIdx.y * 16;
  Acc2 acc = wmma_rowrow2(X, DD, Wm, DD, DD, mtile, ntile, DD);
  const int lane = threadIdx.x & 31;
  const int l16 = lane & 15, half = lane >> 4;
  const int n0 = ntile + l16, n1 = ntile + 16 + l16;
  if (n0 < DD) {
    const float bias = bb[n0];
#pragma unroll
    for (int r = 0; r < 8; ++r)              // C/D layout: VGPR r -> row half*8+r
      outp[(size_t)(mtile + half * 8 + r) * DD + n0] = acc.a0[r] + bias;
  }
  if (n1 < DD) {
    const float bias = bb[n1];
#pragma unroll
    for (int r = 0; r < 8; ++r)
      outp[(size_t)(mtile + half * 8 + r) * DD + n1] = acc.a1[r] + bias;
  }
}

// ================= kernel 2: S = (Q @ K.T) * scale  (bf16 WMMA) ===============
__global__ __launch_bounds__(256) void scores_gemm(const float* __restrict__ Q,
                                                   const float* __restrict__ Km,
                                                   float* __restrict__ S) {
  const int wv    = threadIdx.x >> 5;
  const int ntile = wv * 32;                 // 8 waves cover all 256 cols
  const int mtile = blockIdx.y * 16;
  Acc2 acc = wmma_rowrow2(Q, DD, Km, DD, DD, mtile, ntile, MM);
  const float scale = rsqrtf((float)DD);
  const int lane = threadIdx.x & 31;
  const int l16 = lane & 15, half = lane >> 4;
  const int n0 = ntile + l16, n1 = ntile + 16 + l16;
#pragma unroll
  for (int r = 0; r < 8; ++r) {
    const int m = mtile + half * 8 + r;
    S[(size_t)m * MM + n0] = acc.a0[r] * scale;
    S[(size_t)m * MM + n1] = acc.a1[r] * scale;
  }
}

// ================= kernel 3: row softmax of S [256x256] =======================
__global__ __launch_bounds__(256) void softmax_rows(float* __restrict__ S) {
  __shared__ float red[256];
  const int row = blockIdx.x, t = threadIdx.x;
  const float x = S[(size_t)row * MM + t];
  red[t] = x; __syncthreads();
  for (int s = 128; s; s >>= 1) { if (t < s) red[t] = fmaxf(red[t], red[t + s]); __syncthreads(); }
  const float m = red[0]; __syncthreads();
  const float e = __expf(x - m);
  red[t] = e; __syncthreads();
  for (int s = 128; s; s >>= 1) { if (t < s) red[t] += red[t + s]; __syncthreads(); }
  S[(size_t)row * MM + t] = e / red[0];
}

// ============ kernel 4: w[j] = mean_i attn[i,j] ===============================
__global__ __launch_bounds__(256) void colmean(const float* __restrict__ S,
                                               float* __restrict__ w) {
  const int t = threadIdx.x;
  float acc = 0.f;
  for (int i = 0; i < MM; ++i) acc += S[(size_t)i * MM + t];
  w[t] = acc * (1.0f / MM);
}

// ============ kernel 5: context = w @ V  [D] ==================================
// Stage w[] into LDS with the gfx1250 async global->LDS path (ASYNCcnt).
__global__ __launch_bounds__(256) void ctx_kernel(const float* __restrict__ w,
                                                  const float* __restrict__ V,
                                                  float* __restrict__ ctx) {
  __shared__ float wl[256];
  const int t = threadIdx.x;
  {
    const unsigned ldsa = (unsigned)(uintptr_t)&wl[t];
    const float* gp = w + t;
    asm volatile("global_load_async_to_lds_b32 %0, %1, off"
                 :: "v"(ldsa), "v"(gp) : "memory");
    asm volatile("s_wait_asynccnt 0x0" ::: "memory");
  }
  __syncthreads();
  const int d = blockIdx.x * 256 + t;
  if (d < DD) {
    float acc = 0.f;
    for (int i = 0; i < MM; ++i) acc += wl[i] * V[(size_t)i * DD + d];
    ctx[d] = acc;
  }
}

// ===== kernel 6: extract only the 260 needed columns of W = mu + sigma*eps ====
__global__ __launch_bounds__(256) void extract_cols(const float* __restrict__ mu,
                                                    const float* __restrict__ sg,
                                                    const float* __restrict__ ep,
                                                    float* __restrict__ wcol) {
  const int i = blockIdx.x * 256 + threadIdx.x;
  if (i >= NTOT) return;
  const size_t rowoff = (size_t)i * NTOT + DD;
  for (int c = 0; c < HO; ++c) {
    const float w = fmaf(sg[rowoff + c], ep[rowoff + c], mu[rowoff + c]);
    wcol[(size_t)c * NTOT + i] = w;
  }
}

// ---- TDM: DMA one contiguous column (n fp32) from global into LDS ------------
__device__ __forceinline__ void tdm_load_col(const float* __restrict__ gsrc,
                                             unsigned lds_off, unsigned n) {
  const uint64_t ga = (uint64_t)(uintptr_t)gsrc;
  v4u g0;
  g0[0] = 1u;                                            // count=1, user D#
  g0[1] = lds_off;                                       // lds_addr (bytes)
  g0[2] = (unsigned)ga;                                  // global_addr[31:0]
  g0[3] = ((unsigned)(ga >> 32) & 0x01ffffffu) | 0x80000000u; // addr[56:32]|type=2
  v8i g1;
  g1[0] = (int)(2u << 16);                 // wg_mask=0, data_size=2 (4 bytes)
  g1[1] = (int)((n & 0xffffu) << 16);      // tensor_dim0[15:0]
  g1[2] = (int)((n >> 16) | (1u << 16));   // tensor_dim0[31:16] | tensor_dim1=1
  g1[3] = (int)((n & 0xffffu) << 16);      // tile_dim0 = n
  g1[4] = 1;                               // tile_dim1 = 1
  g1[5] = (int)n;                          // tensor_dim0_stride
  g1[6] = 0;
  g1[7] = 0;
  asm volatile("tensor_load_to_lds %0, %1" :: "s"(g0), "s"(g1) : "memory");
}

// ============ kernel 7: sequential NEAT scan, double-buffered TDM =============
__global__ __launch_bounds__(1024) void scan_kernel(const float* __restrict__ ctx,
                                                    const float* __restrict__ bmu,
                                                    const float* __restrict__ bsg,
                                                    const float* __restrict__ bep,
                                                    const float* __restrict__ wcol,
                                                    float* __restrict__ out) {
  __shared__ float vals[NTOT];
  __shared__ float colb[2][NTOT];
  __shared__ float bvec[HO];
  __shared__ float wsum[32];
  const int t = threadIdx.x;
  for (int i = t; i < NTOT; i += 1024) vals[i] = (i < DD) ? ctx[i] : 0.0f;
  for (int i = t; i < HO; i += 1024)   bvec[i] = fmaf(bsg[i], bep[i], bmu[i]);
  __syncthreads();
  const unsigned c0 = (unsigned)(uintptr_t)&colb[0][0];
  const unsigned c1 = (unsigned)(uintptr_t)&colb[1][0];
  if (t == 0) tdm_load_col(wcol, c0, (unsigned)NTOT);       // prologue prefetch
  for (int jj = 0; jj < HO; ++jj) {
    if (t == 0) __builtin_amdgcn_s_wait_tensorcnt(0);       // buf(jj) ready
    __syncthreads();
    if (t == 0 && jj + 1 < HO)                              // prefetch buf(jj+1)
      tdm_load_col(wcol + (size_t)(jj + 1) * NTOT, (jj & 1) ? c0 : c1,
                   (unsigned)NTOT);
    const float* cb = &colb[jj & 1][0];
    float p = 0.0f;
    for (int i = t; i < NTOT; i += 1024) p += vals[i] * cb[i];
    for (int o = 16; o; o >>= 1) p += __shfl_down(p, o, 32);  // wave32 reduce
    if ((t & 31) == 0) wsum[t >> 5] = p;
    __syncthreads();
    if (t < 32) {
      float q = wsum[t];
      for (int o = 16; o; o >>= 1) q += __shfl_down(q, o, 32);
      if (t == 0) vals[DD + jj] = tanhf(q + bvec[jj]);
    }
    __syncthreads();
  }
  if (t < OO) out[t] = vals[NTOT - OO + t];
}

extern "C" void kernel_launch(void* const* d_in, const int* in_sizes, int n_in,
                              void* d_out, int out_size, void* d_ws, size_t ws_size,
                              hipStream_t stream) {
  (void)in_sizes; (void)n_in; (void)out_size; (void)ws_size;
  const float* X   = (const float*)d_in[0];
  const float* Wq  = (const float*)d_in[1];
  const float* bq  = (const float*)d_in[2];
  const float* Wk  = (const float*)d_in[3];
  const float* bk  = (const float*)d_in[4];
  const float* Wv  = (const float*)d_in[5];
  const float* bv  = (const float*)d_in[6];
  const float* wmu = (const float*)d_in[7];
  const float* wsg = (const float*)d_in[8];
  const float* bmu = (const float*)d_in[9];
  const float* bsg = (const float*)d_in[10];
  const float* epw = (const float*)d_in[11];
  const float* epb = (const float*)d_in[12];

  float* ws = (float*)d_ws;
  size_t off = 0;
  float* Qb   = ws + off; off += (size_t)MM * DD;
  float* Kb   = ws + off; off += (size_t)MM * DD;
  float* Vb   = ws + off; off += (size_t)MM * DD;
  float* Sb   = ws + off; off += (size_t)MM * MM;
  float* wb   = ws + off; off += 256;
  float* ctx  = ws + off; off += 7696;
  float* wcol = ws + off; off += (size_t)HO * NTOT;   // total ~32.2 MB

  extract_cols<<<dim3((NTOT + 255) / 256), dim3(256), 0, stream>>>(wmu, wsg, epw, wcol);
  qkv_gemm<<<dim3((DD + 255) / 256, MM / 16, 3), dim3(256), 0, stream>>>(
      X, Wq, bq, Wk, bk, Wv, bv, Qb, Kb, Vb);
  scores_gemm<<<dim3(1, MM / 16), dim3(256), 0, stream>>>(Qb, Kb, Sb);
  softmax_rows<<<dim3(MM), dim3(256), 0, stream>>>(Sb);
  colmean<<<dim3(1), dim3(256), 0, stream>>>(Sb, wb);
  ctx_kernel<<<dim3((DD + 255) / 256), dim3(256), 0, stream>>>(wb, Vb, ctx);
  scan_kernel<<<dim3(1), dim3(1024), 0, stream>>>(ctx, bmu, bsg, epb, wcol,
                                                  (float*)d_out);
}
